// OneBitLinear_15290083574248
// MI455X (gfx1250) — matrix operand order repaired and checked by reading, not compiled, not run on hardware
//
#include <hip/hip_runtime.h>

// out = ((x * g) @ sign^T) * h + bias
// x:[8192,4096] f32, sign:[4096,4096] f32 (+/-1), g:[4096], h:[4096], bias:[4096]
// fp16 WMMA (v_wmma_f32_16x16x32_f16) with f32 accumulation. sign is exact in fp16.

#define N_DIM   8192
#define IN_DIM  4096
#define OUT_DIM 4096

#define BM 128   // rows of x per block
#define BN 128   // output columns (rows of sign) per block
#define BK 32    // K step = WMMA K
#define LDK 40   // padded LDS K stride (halves): rows 80B -> 16B-aligned chunks

typedef _Float16 v16h __attribute__((ext_vector_type(16)));
typedef _Float16 v8h  __attribute__((ext_vector_type(8)));
typedef _Float16 v4h  __attribute__((ext_vector_type(4)));
typedef float    v8f  __attribute__((ext_vector_type(8)));

__global__ __launch_bounds__(256) void onebit_gemm_wmma(
    const float* __restrict__ x,
    const float* __restrict__ sgn,
    const float* __restrict__ g,
    const float* __restrict__ h,
    const float* __restrict__ bias,
    float* __restrict__ out)
{
    __shared__ _Float16 As[BM][LDK];
    __shared__ _Float16 Bs[BN][LDK];

    const int tid  = threadIdx.x;
    const int lane = tid & 31;
    const int wid  = tid >> 5;
    const int waveM = wid & 3;   // 4 wave-rows of 32
    const int waveN = wid >> 2;  // 2 wave-cols of 64

    const int mblk = blockIdx.y * BM;   // x row base
    const int oblk = blockIdx.x * BN;   // output col base (sign row base)

    // Global staging assignment: 256 threads, each loads 4 chunks of 4 floats
    // for A and B per K-step (128x32 tile each).
    const int cg = tid & 7;    // column group (4 floats each)
    const int r0 = tid >> 3;   // base row 0..31 (strided by 32, 4 reps)

    const float* xA = x   + (size_t)(mblk + r0) * IN_DIM + cg * 4;
    const float* sB = sgn + (size_t)(oblk + r0) * IN_DIM + cg * 4;

    v8f acc[2][4];
    #pragma unroll
    for (int i = 0; i < 2; ++i)
        #pragma unroll
        for (int j = 0; j < 4; ++j)
            acc[i][j] = (v8f){};

    // Fragment addressing (per ISA 16-bit layouts, wave32):
    // A: lane holds row M=lane%16; half-chunks at K = 8*(lane/16) and 16+8*(lane/16)
    // B: lane holds col N=lane%16; half-chunks at K = 8*(lane/16) and 16+8*(lane/16)
    const int lhalf = lane >> 4;        // 0 or 1
    const int lrow  = lane & 15;
    const int kbA   = lhalf * 8;
    const int kbB   = lhalf * 8;

    for (int k = 0; k < IN_DIM; k += BK) {
        // ---- stage global -> registers (fp32), convert to fp16 ----
        const float4 gv = *(const float4*)(g + k + cg * 4);
        float4 av[4], bv[4];
        #pragma unroll
        for (int r = 0; r < 4; ++r) {
            av[r] = *(const float4*)(xA + k + (size_t)r * 32 * IN_DIM);
            bv[r] = *(const float4*)(sB + k + (size_t)r * 32 * IN_DIM);
        }

        __syncthreads();   // previous iteration's LDS reads complete

        #pragma unroll
        for (int r = 0; r < 4; ++r) {
            v4h a16, b16;
            a16[0] = (_Float16)(av[r].x * gv.x);
            a16[1] = (_Float16)(av[r].y * gv.y);
            a16[2] = (_Float16)(av[r].z * gv.z);
            a16[3] = (_Float16)(av[r].w * gv.w);
            b16[0] = (_Float16)bv[r].x;
            b16[1] = (_Float16)bv[r].y;
            b16[2] = (_Float16)bv[r].z;
            b16[3] = (_Float16)bv[r].w;
            *(v4h*)&As[r0 + 32 * r][cg * 4] = a16;   // 8B store, 8B aligned
            *(v4h*)&Bs[r0 + 32 * r][cg * 4] = b16;
        }

        __syncthreads();   // tiles visible to all waves

        // ---- load fragments from LDS ----
        v16h afrag[2], bfrag[4];
        #pragma unroll
        for (int f = 0; f < 2; ++f) {
            const int row = waveM * 32 + f * 16 + lrow;
            v8h lo = *(const v8h*)&As[row][kbA];        // 16B aligned
            v8h hi = *(const v8h*)&As[row][16 + kbA];
            afrag[f] = __builtin_shufflevector(lo, hi,
                0,1,2,3,4,5,6,7,8,9,10,11,12,13,14,15);
        }
        #pragma unroll
        for (int j = 0; j < 4; ++j) {
            const int row = waveN * 64 + j * 16 + lrow;
            v8h lo = *(const v8h*)&Bs[row][kbB];
            v8h hi = *(const v8h*)&Bs[row][16 + kbB];
            bfrag[j] = __builtin_shufflevector(lo, hi,
                0,1,2,3,4,5,6,7,8,9,10,11,12,13,14,15);
        }

        // ---- 8 WMMAs: 32x64 output tile per wave ----
        #pragma unroll
        for (int f = 0; f < 2; ++f)
            #pragma unroll
            for (int j = 0; j < 4; ++j)
                acc[f][j] = __builtin_amdgcn_wmma_f32_16x16x32_f16(
                    false, afrag[f], false, bfrag[j],
                    (short)0, acc[f][j], false, false);
    }

    // ---- epilogue: D layout => col o = lane%16, row = v + 8*(lane/16) ----
    #pragma unroll
    for (int j = 0; j < 4; ++j) {
        const int o  = oblk + waveN * 64 + j * 16 + lrow;
        const float hv = h[o];
        const float bv = bias[o];
        #pragma unroll
        for (int f = 0; f < 2; ++f) {
            const int rbase = mblk + waveM * 32 + f * 16 + lhalf * 8;
            #pragma unroll
            for (int v = 0; v < 8; ++v) {
                out[(size_t)(rbase + v) * OUT_DIM + o] = acc[f][j][v] * hv + bv;
            }
        }
    }
}

extern "C" void kernel_launch(void* const* d_in, const int* in_sizes, int n_in,
                              void* d_out, int out_size, void* d_ws, size_t ws_size,
                              hipStream_t stream) {
    (void)in_sizes; (void)n_in; (void)d_ws; (void)ws_size; (void)out_size;
    const float* x    = (const float*)d_in[0];
    const float* sgn  = (const float*)d_in[1];
    const float* g    = (const float*)d_in[2];
    const float* h    = (const float*)d_in[3];
    const float* bias = (const float*)d_in[4];
    float* out = (float*)d_out;

    dim3 grid(OUT_DIM / BN, N_DIM / BM);   // 32 x 64
    dim3 block(256);
    onebit_gemm_wmma<<<grid, block, 0, stream>>>(x, sgn, g, h, bias, out);
}
